// MultiHeadSelfAttention_68771016344236
// MI455X (gfx1250) — matrix operand order repaired and checked by reading, not compiled, not run on hardware
//
#include <hip/hip_runtime.h>
#include <hip/hip_bf16.h>

// ---------------------------------------------------------------------------
// MHSA for MI455X (gfx1250): bf16 WMMA (v_wmma_f32_16x16x32_bf16), wave32,
// async global->LDS staging (ASYNCcnt) for GEMM operands.
// B=32, S=1024, D=512, H=8, DK=64.
// ---------------------------------------------------------------------------

#define BB 32
#define SS 1024
#define DD 512
#define HH 8
#define DKK 64

typedef __attribute__((ext_vector_type(16))) __bf16 v16bf;
typedef __attribute__((ext_vector_type(8)))  float  v8f;
typedef __attribute__((ext_vector_type(4)))  int    v4i;

// pointer-to-int4 in global (AS1) / LDS (AS3) address spaces
typedef __attribute__((address_space(1))) v4i* gas_v4i_ptr;
typedef __attribute__((address_space(3))) v4i* las_v4i_ptr;

__device__ __forceinline__ v8f wmma_bf16(v16bf a, v16bf b, v8f c) {
  // 8 args: (neg_a, A, neg_b, B, c_mod, C, reuse_a, reuse_b)
  return __builtin_amdgcn_wmma_f32_16x16x32_bf16(false, a, false, b, (short)0, c,
                                                 false, false);
}

// A-fragment (16x32 bf16): lane = half*16+row; element e -> k index.
__device__ __forceinline__ int a_kidx(int e, int half) {
  return e + (half << 4);
}

// ---- async global->LDS copy (gfx1250 GLOBAL_LOAD_ASYNC_TO_LDS_B128) -------
#if defined(__has_builtin)
#if __has_builtin(__builtin_amdgcn_global_load_async_to_lds_b128)
#define HAS_ASYNC_LDS 1
#endif
#endif

__device__ __forceinline__ void async_copy16(const float* gsrc, float* ldst) {
#ifdef HAS_ASYNC_LDS
  __builtin_amdgcn_global_load_async_to_lds_b128(
      (gas_v4i_ptr)(v4i*)gsrc, (las_v4i_ptr)(v4i*)ldst, 0, 0);
#else
  *(float4*)ldst = *(const float4*)gsrc;
#endif
}

__device__ __forceinline__ void async_wait0() {
#ifdef HAS_ASYNC_LDS
#if __has_builtin(__builtin_amdgcn_s_wait_asynccnt)
  __builtin_amdgcn_s_wait_asynccnt(0);
#else
  asm volatile("s_wait_asynccnt 0x0" ::: "memory");
#endif
#endif
}

// ---------------------------------------------------------------------------
// Kernel 0: zero the (B,D) mean-output region of d_out.
// ---------------------------------------------------------------------------
__global__ void mhsa_zero_kernel(float* out, int n) {
  int i = blockIdx.x * blockDim.x + threadIdx.x;
  if (i < n) out[i] = 0.0f;
}

// ---------------------------------------------------------------------------
// Kernel 1: QKV projections.  Y = z @ W + bias for W in {Wq,Wk,Wv}.
// grid = (M/16, D/128), block = 256 (8 waves).  Each wave owns one 16-col
// tile and accumulates Q,K,V simultaneously.  The 16x32 z tile and the three
// 32x128 W slabs are staged fp32 in LDS via async b128 copies per k-step;
// fp32 -> bf16 conversion happens only when fragments are built.
// ---------------------------------------------------------------------------
#define ZP 36    // z LDS pitch (floats): 16B-aligned rows, spreads banks
#define WP 132   // W LDS pitch (floats): 16B-aligned rows

__global__ void mhsa_qkv_kernel(const float* __restrict__ z,
                                const float* __restrict__ Wq, const float* __restrict__ bq,
                                const float* __restrict__ Wk, const float* __restrict__ bk,
                                const float* __restrict__ Wv, const float* __restrict__ bv,
                                __bf16* __restrict__ Qb, __bf16* __restrict__ Kb,
                                __bf16* __restrict__ Vb) {
  __shared__ float ztf[16 * ZP];          // 16 x 32 fp32 z tile
  __shared__ float wt[3 * 32 * WP];       // 3 matrices x (32k x 128n) fp32

  const int tid  = threadIdx.x;
  const int wave = tid >> 5;
  const int lane = tid & 31;
  const int half = lane >> 4;
  const int ln   = lane & 15;
  const int m0   = blockIdx.x * 16;              // row tile in [0, B*S)
  const int nb   = blockIdx.y * 128;             // block col base
  const int n0   = nb + wave * 16;               // col tile for this wave

  const float* Ws[3] = {Wq, Wk, Wv};
  const float* bs[3] = {bq, bk, bv};
  __bf16*      Os[3] = {Qb, Kb, Vb};

  v8f acc[3] = {};

  for (int k0 = 0; k0 < DD; k0 += 32) {
    __syncthreads();  // previous iteration's LDS reads complete

    // ---- async-stage z tile: 16 rows x 32 floats = 128 b128 transfers ----
    if (tid < 128) {
      int r  = tid >> 3;
      int c4 = (tid & 7) * 4;
      async_copy16(&z[(size_t)(m0 + r) * DD + k0 + c4], &ztf[r * ZP + c4]);
    }
    // ---- async-stage W slabs: 3 x 32 x 128 floats = 3072 b128 transfers --
#pragma unroll
    for (int it = 0; it < 12; ++it) {
      int idx = tid + it * 256;          // [0, 3072)
      int m   = idx >> 10;               // matrix
      int rem = idx & 1023;
      int kr  = rem >> 5;                // k row within slab
      int c4  = (rem & 31) * 4;          // float4 col
      async_copy16(&Ws[m][(size_t)(k0 + kr) * DD + nb + c4],
                   &wt[(m * 32 + kr) * WP + c4]);
    }
    async_wait0();
    __syncthreads();

    // A fragment: shared z tile (fp32 -> bf16), 16-bit 16x32 layout
    v16bf a;
#pragma unroll
    for (int e = 0; e < 16; ++e) a[e] = (__bf16)ztf[ln * ZP + a_kidx(e, half)];

#pragma unroll
    for (int m = 0; m < 3; ++m) {
      v16bf bf_;
#pragma unroll
      for (int e = 0; e < 16; ++e) {
        int kr = half * 16 + e;  // B fragment: 32x16, krow = half*16+e, col = ln
        bf_[e] = (__bf16)wt[(m * 32 + kr) * WP + wave * 16 + ln];
      }
      acc[m] = wmma_bf16(a, bf_, acc[m]);
    }
  }

  // epilogue: + bias, store bf16 (C layout: VGPR i -> row i + half*8, col ln)
#pragma unroll
  for (int m = 0; m < 3; ++m) {
    float bias = bs[m][n0 + ln];
#pragma unroll
    for (int i = 0; i < 8; ++i) {
      int row = m0 + i + half * 8;
      Os[m][(size_t)row * DD + n0 + ln] = (__bf16)(acc[m][i] + bias);
    }
  }
}

// ---------------------------------------------------------------------------
// Kernel 2: attention.  One block per (b, 16-row tile); loops all 8 heads so
// the head-mean of attn accumulates in LDS and is written exactly once.
// LDS: scores[16][SPITCH] f32 + mean[16][SPITCH] f32 + partial[4][16][16] f32.
// ---------------------------------------------------------------------------
#define SPITCH 1028  // 1024 + 4: rows land on distinct bank groups

__global__ void mhsa_attn_kernel(const __bf16* __restrict__ Qb,
                                 const __bf16* __restrict__ Kb,
                                 const __bf16* __restrict__ Vb,
                                 __bf16* __restrict__ Cb,
                                 float* __restrict__ attn_out) {
  extern __shared__ char smem[];
  float* sc   = (float*)smem;            // [16][SPITCH] scores/attn
  float* mn   = sc + 16 * SPITCH;        // [16][SPITCH] mean over heads
  float* part = mn + 16 * SPITCH;        // [4][16][16] partial C

  const int tid  = threadIdx.x;
  const int wave = tid >> 5;
  const int lane = tid & 31;
  const int half = lane >> 4;
  const int ln   = lane & 15;
  const int b    = blockIdx.x >> 6;  // 64 row-tiles per batch
  const int rt   = blockIdx.x & 63;
  const size_t baseS = (size_t)b * SS;            // sequence base (rows of S)
  const size_t baseQ = (baseS + rt * 16) * DD;    // this block's Q rows
  const float  scale = 0.125f;                    // 1/sqrt(DK)

  for (int h = 0; h < HH; ++h) {
    const int d0 = h * DKK;

    // ---- scores = (Q K^T) * scale : each wave does 8 key col-tiles ----
    v16bf aq[2];
#pragma unroll
    for (int kk = 0; kk < 2; ++kk)
#pragma unroll
      for (int e = 0; e < 16; ++e)
        aq[kk][e] = Qb[baseQ + (size_t)ln * DD + d0 + kk * 32 + a_kidx(e, half)];

    for (int it = 0; it < 8; ++it) {
      int ct = wave * 8 + it;
      int t0 = ct * 16;
      v8f c = {};
#pragma unroll
      for (int kk = 0; kk < 2; ++kk) {
        v16bf bk_;  // B = K^T: B[kdim][n] = K[t0+n][d0+kdim]
#pragma unroll
        for (int e = 0; e < 16; ++e) {
          int kr = half * 16 + e;
          bk_[e] = Kb[(baseS + t0 + ln) * DD + d0 + kk * 32 + kr];
        }
        c = wmma_bf16(aq[kk], bk_, c);
      }
#pragma unroll
      for (int i = 0; i < 8; ++i)
        sc[(i + half * 8) * SPITCH + t0 + ln] = c[i] * scale;
    }
    __syncthreads();

    // ---- softmax over each of the 16 rows (2 rows per wave, wave32) ----
    for (int rr = 0; rr < 2; ++rr) {
      int r = wave * 2 + rr;
      float* row = sc + r * SPITCH;
      float mx = -3.4e38f;
      for (int c0 = lane; c0 < SS; c0 += 32) mx = fmaxf(mx, row[c0]);
#pragma unroll
      for (int off = 16; off > 0; off >>= 1) mx = fmaxf(mx, __shfl_xor(mx, off, 32));
      float sum = 0.f;
      for (int c0 = lane; c0 < SS; c0 += 32) {
        float e_ = __expf(row[c0] - mx);
        row[c0]  = e_;
        sum += e_;
      }
#pragma unroll
      for (int off = 16; off > 0; off >>= 1) sum += __shfl_xor(sum, off, 32);
      float inv = 1.f / sum;
      float* mrow = mn + r * SPITCH;
      for (int c0 = lane; c0 < SS; c0 += 32) {
        float a_ = row[c0] * inv;
        row[c0]  = a_;
        float contrib = a_ * (1.0f / HH);
        mrow[c0] = (h == 0) ? contrib : (mrow[c0] + contrib);
      }
    }
    __syncthreads();

    // ---- context = attn @ V : 4 col-tiles x 2 K-halves over 8 waves ----
    int ctile = wave & 3;
    int khalf = wave >> 2;
    v8f cacc = {};
    for (int kt = 0; kt < 16; ++kt) {
      int kbase = khalf * 512 + kt * 32;
      v16bf aa;  // attn tile from LDS, converted fp32 -> bf16
#pragma unroll
      for (int e = 0; e < 16; ++e)
        aa[e] = (__bf16)sc[ln * SPITCH + kbase + a_kidx(e, half)];
      v16bf bv_;
#pragma unroll
      for (int e = 0; e < 16; ++e) {
        int kr = half * 16 + e;
        bv_[e] = Vb[(baseS + kbase + kr) * DD + d0 + ctile * 16 + ln];
      }
      cacc = wmma_bf16(aa, bv_, cacc);
    }
    if (khalf == 1) {
#pragma unroll
      for (int i = 0; i < 8; ++i)
        part[ctile * 256 + (i + half * 8) * 16 + ln] = cacc[i];
    }
    __syncthreads();
    if (khalf == 0) {
#pragma unroll
      for (int i = 0; i < 8; ++i) {
        float v = cacc[i] + part[ctile * 256 + (i + half * 8) * 16 + ln];
        int row = rt * 16 + i + half * 8;
        Cb[(baseS + row) * DD + d0 + ctile * 16 + ln] = (__bf16)v;
      }
    }
    __syncthreads();  // protect sc/part reuse in next head
  }

  // ---- write head-mean attention tile: d_out[B*D + b*S*S + ...] ----
  float* dst = attn_out + (size_t)b * SS * SS + (size_t)(rt * 16) * SS;
  for (int i = tid; i < 16 * SS; i += 256) {
    int r = i >> 10, c = i & (SS - 1);
    dst[(size_t)r * SS + c] = mn[r * SPITCH + c];
  }
}

// ---------------------------------------------------------------------------
// Kernel 3: x = context @ Wf + bf + z; per-row LayerNorm; atomic-accumulate
// x/S into the (B,D) mean output.  One block per 16-row panel (full 512 cols).
// ---------------------------------------------------------------------------
#define XPITCH 520

__global__ void mhsa_final_kernel(const __bf16* __restrict__ Cb,
                                  const float* __restrict__ Wf,
                                  const float* __restrict__ bf,
                                  const float* __restrict__ z,
                                  const float* __restrict__ gamma,
                                  const float* __restrict__ beta,
                                  float* __restrict__ out_mean) {
  __shared__ float xs[16 * XPITCH];  // 16 x 512 fp32 panel

  const int tid  = threadIdx.x;
  const int wave = tid >> 5;
  const int lane = tid & 31;
  const int half = lane >> 4;
  const int ln   = lane & 15;
  const int m0   = blockIdx.x * 16;
  const int b    = m0 / SS;

  v8f acc[4] = {};
  for (int k0 = 0; k0 < DD; k0 += 32) {
    v16bf a;
#pragma unroll
    for (int e = 0; e < 16; ++e)
      a[e] = Cb[(size_t)(m0 + ln) * DD + k0 + a_kidx(e, half)];
#pragma unroll
    for (int j = 0; j < 4; ++j) {
      int n0 = wave * 64 + j * 16;
      v16bf bfr;
#pragma unroll
      for (int e = 0; e < 16; ++e) {
        int kr = half * 16 + e;
        bfr[e] = (__bf16)Wf[(size_t)(k0 + kr) * DD + n0 + ln];
      }
      acc[j] = wmma_bf16(a, bfr, acc[j]);
    }
  }

#pragma unroll
  for (int j = 0; j < 4; ++j) {
    int n0  = wave * 64 + j * 16;
    int col = n0 + ln;
    float bias = bf[col];
#pragma unroll
    for (int i = 0; i < 8; ++i) {
      int rloc = i + half * 8;
      float v  = acc[j][i] + bias + z[(size_t)(m0 + rloc) * DD + col];
      xs[rloc * XPITCH + col] = v;
    }
  }
  __syncthreads();

  // LayerNorm per row (2 rows per wave), then mean over S via atomics
  for (int rr = 0; rr < 2; ++rr) {
    int r = wave * 2 + rr;
    float* row = xs + r * XPITCH;
    float s_ = 0.f;
    for (int c = lane; c < DD; c += 32) s_ += row[c];
#pragma unroll
    for (int off = 16; off > 0; off >>= 1) s_ += __shfl_xor(s_, off, 32);
    float mu = s_ * (1.0f / DD);
    float v_ = 0.f;
    for (int c = lane; c < DD; c += 32) {
      float d = row[c] - mu;
      v_ += d * d;
    }
#pragma unroll
    for (int off = 16; off > 0; off >>= 1) v_ += __shfl_xor(v_, off, 32);
    float rstd = rsqrtf(v_ * (1.0f / DD) + 1e-5f);
    for (int c = lane; c < DD; c += 32) {
      float xl = (row[c] - mu) * rstd * gamma[c] + beta[c];
      atomicAdd(&out_mean[(size_t)b * DD + c], xl * (1.0f / SS));
    }
  }
}

// ---------------------------------------------------------------------------
// Host launcher
// ---------------------------------------------------------------------------
extern "C" void kernel_launch(void* const* d_in, const int* in_sizes, int n_in,
                              void* d_out, int out_size, void* d_ws, size_t ws_size,
                              hipStream_t stream) {
  const float* z     = (const float*)d_in[0];
  const float* Wq    = (const float*)d_in[1];
  const float* bq    = (const float*)d_in[2];
  const float* Wk    = (const float*)d_in[3];
  const float* bk    = (const float*)d_in[4];
  const float* Wv    = (const float*)d_in[5];
  const float* bv    = (const float*)d_in[6];
  const float* Wf    = (const float*)d_in[7];
  const float* bf    = (const float*)d_in[8];
  const float* gamma = (const float*)d_in[9];
  const float* beta  = (const float*)d_in[10];

  float* out      = (float*)d_out;
  float* attn_out = out + (size_t)BB * DD;  // second tuple element

  const size_t nZ = (size_t)BB * SS * DD;   // 16,777,216
  __bf16* Qb = (__bf16*)d_ws;
  __bf16* Kb = Qb + nZ;
  __bf16* Vb = Kb + nZ;
  __bf16* Cb = Vb + nZ;

  // 0) zero the (B,D) mean region
  mhsa_zero_kernel<<<(BB * DD + 255) / 256, 256, 0, stream>>>(out, BB * DD);

  // 1) QKV projections: grid (M/16, D/128)
  dim3 g1(BB * SS / 16, DD / 128);
  mhsa_qkv_kernel<<<g1, 256, 0, stream>>>(z, Wq, bq, Wk, bk, Wv, bv, Qb, Kb, Vb);

  // 2) attention: one block per (b, 16-row tile); dynamic LDS ~132 KB
  size_t smem = (size_t)(2 * 16 * SPITCH + 4 * 16 * 16) * sizeof(float);
  mhsa_attn_kernel<<<BB * (SS / 16), 256, smem, stream>>>(Qb, Kb, Vb, Cb, attn_out);

  // 3) final projection + residual + LN + mean over S
  mhsa_final_kernel<<<BB * SS / 16, 256, 0, stream>>>(Cb, Wf, bf, z, gamma, beta, out);
}
